// GCN_21285857919396
// MI455X (gfx1250) — matrix operand order repaired and checked by reading, not compiled, not run on hardware
//
#include <hip/hip_runtime.h>
#include <hip/hip_bf16.h>

typedef __attribute__((ext_vector_type(16))) _Float16 v16h;
typedef __attribute__((ext_vector_type(8)))  _Float16 v8h;
typedef __attribute__((ext_vector_type(4)))  _Float16 v4h;
typedef __attribute__((ext_vector_type(8)))  float    v8f;

#define HID 128

// ---------------- utility kernels ----------------

__global__ void gcn_zero_f32(float* __restrict__ p, long n) {
    long i = (long)blockIdx.x * blockDim.x + threadIdx.x;
    if (i < n) p[i] = 0.0f;
}

// b128 zero fill for the [N x 128] buffers (count in float4 units)
__global__ void gcn_zero_f32x4(float4* __restrict__ p, long n4) {
    long i = (long)blockIdx.x * blockDim.x + threadIdx.x;
    if (i < n4) p[i] = make_float4(0.f, 0.f, 0.f, 0.f);
}

__global__ void gcn_deg_count(const int* __restrict__ dst, int E, float* __restrict__ deg) {
    int e = blockIdx.x * blockDim.x + threadIdx.x;
    if (e < E) atomicAdd(&deg[dst[e]], 1.0f);
}

__global__ void gcn_dinv(float* __restrict__ dinv, long n) {
    long i = (long)blockIdx.x * blockDim.x + threadIdx.x;
    if (i < n) dinv[i] = rsqrtf(dinv[i] + 1.0f);   // deg + self-loop, then ^-0.5
}

// fp32 -> fp16 activation copy, 4 elements/thread (b128 load, b64 store)
__global__ void gcn_cvt_f16(const float* __restrict__ s, _Float16* __restrict__ d, long n4) {
    long i = (long)blockIdx.x * blockDim.x + threadIdx.x;
    if (i >= n4) return;
    const float4 v = *(const float4*)(s + i * 4);
    v4h o; o.x = (_Float16)v.x; o.y = (_Float16)v.y; o.z = (_Float16)v.z; o.w = (_Float16)v.w;
    *(v4h*)(d + i * 4) = o;
}

// W[k][n] (128x128 f32) -> Wt[n][k] (f16), so B fragments load contiguously
__global__ void gcn_wt_cvt(const float* __restrict__ W, _Float16* __restrict__ Wt) {
    int k = blockIdx.x;      // 128 blocks
    int n = threadIdx.x;     // 128 threads
    Wt[n * HID + k] = (_Float16)W[k * HID + n];
}

// ---------------- WMMA GEMM: C[M x 128] = A[M x 128] * W ----------------
// A: f16 row-major. Bt: f16 [128 x 128] = W^T (row n holds column n of W over k).
// One block per 16-row M tile; 8 waves cover the 8 N-tiles. K=128 -> 4 WMMA steps.
__global__ __launch_bounds__(256)
void gcn_gemm_wmma(const _Float16* __restrict__ A, const _Float16* __restrict__ Bt,
                   float* __restrict__ C, int M) {
    const int wave = threadIdx.x >> 5;   // tileN 0..7
    const int lane = threadIdx.x & 31;
    const int half = lane >> 4;          // K half-select per ISA fragment layout
    const int r    = lane & 15;
    const int tileM = blockIdx.x;

    int row = tileM * 16 + r;
    if (row >= M) row = M - 1;           // clamp loads; keep EXEC all-ones for WMMA
    const _Float16* arow = A  + (long)row * HID;
    const _Float16* brow = Bt + (long)(wave * 16 + r) * HID;

    v8f c = {};
#pragma unroll
    for (int kk = 0; kk < 128; kk += 32) {
        // A 16x32 f16 fragment: lane half=0 -> K {kk..kk+7, kk+16..kk+23}, half=1 -> +8
        v8h a_lo = *(const v8h*)(arow + kk + half * 8);
        v8h a_hi = *(const v8h*)(arow + kk + 16 + half * 8);
        v16h a = __builtin_shufflevector(a_lo, a_hi, 0,1,2,3,4,5,6,7,8,9,10,11,12,13,14,15);
        // B 32x16 fragment: lane col = r, K {kk..kk+7, kk+16..kk+23} + half*8 (same as A)
        v8h b_lo = *(const v8h*)(brow + kk + half * 8);
        v8h b_hi = *(const v8h*)(brow + kk + 16 + half * 8);
        v16h b = __builtin_shufflevector(b_lo, b_hi, 0,1,2,3,4,5,6,7,8,9,10,11,12,13,14,15);
        c = __builtin_amdgcn_wmma_f32_16x16x32_f16(false, a, false, b, (short)0, c,
                                                   false, false);
    }

    // C/D layout: VGPR v -> row tileM*16 + v + 8*half, col wave*16 + r
    float* cbase = C + (long)(tileM * 16 + 8 * half) * HID + wave * 16 + r;
    if (tileM * 16 + 16 <= M) {
        // full tile (always true when M % 16 == 0): unguarded, clause-able stores
#pragma unroll
        for (int v = 0; v < 8; ++v) cbase[(long)v * HID] = c[v];
    } else {
        // ragged tail tile: per-row predicated stores
#pragma unroll
        for (int v = 0; v < 8; ++v) {
            int orow = tileM * 16 + 8 * half + v;
            if (orow < M) cbase[(long)v * HID] = c[v];
        }
    }
}

// ---------------- edge scatter: agg[dst] += h[src] * dinv[src]*dinv[dst] ----------------
// 32 lanes per edge, float4 per lane -> 128 feats. Atomics resolve in L2 (h fits in 192MB L2).
__global__ __launch_bounds__(256)
void gcn_scatter(const float* __restrict__ h, const int* __restrict__ src,
                 const int* __restrict__ dst, const float* __restrict__ dinv,
                 float* __restrict__ agg, int E) {
    long t = (long)blockIdx.x * blockDim.x + threadIdx.x;
    int e = (int)(t >> 5);
    if (e >= E) return;
    int f = ((int)t & 31) * 4;
    int s = src[e], d = dst[e];
    float norm = dinv[s] * dinv[d];
    const float4 hv = *(const float4*)(h + (long)s * HID + f);
    float* ap = agg + (long)d * HID + f;
    atomicAdd(ap + 0, hv.x * norm);
    atomicAdd(ap + 1, hv.y * norm);
    atomicAdd(ap + 2, hv.z * norm);
    atomicAdd(ap + 3, hv.w * norm);
}

// out = [relu](agg + h * dinv^2 + bias), 4 feats/thread, in-place on agg (elementwise)
__global__ void gcn_finalize(float* __restrict__ agg, const float* __restrict__ h,
                             const float* __restrict__ dinv, const float* __restrict__ bias,
                             int relu, long nh4) {
    long i = (long)blockIdx.x * blockDim.x + threadIdx.x;   // float4 index
    if (i >= nh4) return;
    long node = i >> 5;               // 32 float4 per node row
    int  f4   = (int)(i & 31) * 4;
    float di = dinv[node];
    float d2 = di * di;
    float4 a = *(const float4*)(agg + i * 4);
    float4 hv = *(const float4*)(h + i * 4);
    float4 b = *(const float4*)(bias + f4);
    float4 v;
    v.x = a.x + hv.x * d2 + b.x;
    v.y = a.y + hv.y * d2 + b.y;
    v.z = a.z + hv.z * d2 + b.z;
    v.w = a.w + hv.w * d2 + b.w;
    if (relu) {
        v.x = fmaxf(v.x, 0.f); v.y = fmaxf(v.y, 0.f);
        v.z = fmaxf(v.z, 0.f); v.w = fmaxf(v.w, 0.f);
    }
    *(float4*)(agg + i * 4) = v;
}

// ---------------- decode: sigmoid(dot(z[src], z[dst])) per edge, wave32 reduction ----------------
__global__ __launch_bounds__(256)
void gcn_decode(const float* __restrict__ z, const int* __restrict__ src,
                const int* __restrict__ dst, float* __restrict__ out, int E) {
    long t = (long)blockIdx.x * blockDim.x + threadIdx.x;
    int e = (int)(t >> 5);
    if (e >= E) return;
    int lane = (int)t & 31;
    int s = src[e], d = dst[e];
    const float4 a = *(const float4*)(z + (long)s * HID + lane * 4);
    const float4 b = *(const float4*)(z + (long)d * HID + lane * 4);
    float v = a.x * b.x + a.y * b.y + a.z * b.z + a.w * b.w;
#pragma unroll
    for (int off = 16; off > 0; off >>= 1) v += __shfl_xor(v, off, 32);
    if (lane == 0) out[e] = 1.0f / (1.0f + __expf(-v));
}

// ---------------- launch ----------------

extern "C" void kernel_launch(void* const* d_in, const int* in_sizes, int n_in,
                              void* d_out, int out_size, void* d_ws, size_t ws_size,
                              hipStream_t stream) {
    const float* x  = (const float*)d_in[0];
    const float* Ws[3] = { (const float*)d_in[1], (const float*)d_in[3], (const float*)d_in[5] };
    const float* bs[3] = { (const float*)d_in[2], (const float*)d_in[4], (const float*)d_in[6] };
    const int*   ei = (const int*)d_in[7];

    const long N  = (long)in_sizes[0] / HID;
    const int  E  = in_sizes[7] / 2;
    const int* src = ei;
    const int* dstp = ei + E;
    const long NH = N * HID;

    // workspace carve-up (256B aligned)
    char* ws = (char*)d_ws;
    size_t off = 0;
    auto carve = [&](size_t bytes) {
        char* p = ws + off;
        off = (off + bytes + 255) & ~(size_t)255;
        return p;
    };
    float*    dinv = (float*)   carve(N * sizeof(float));
    float*    bufA = (float*)   carve(NH * sizeof(float));
    float*    bufB = (float*)   carve(NH * sizeof(float));
    float*    hbuf = (float*)   carve(NH * sizeof(float));
    _Float16* actH = (_Float16*)carve(NH * sizeof(_Float16));
    _Float16* Wt   = (_Float16*)carve((size_t)HID * HID * sizeof(_Float16));
    (void)ws_size; (void)n_in;

    const int B = 256;
    auto blocks = [&](long work) { return (unsigned)((work + B - 1) / B); };

    // degrees -> dinv (recomputed every call: deterministic, ws is not preserved)
    gcn_zero_f32<<<blocks(N), B, 0, stream>>>(dinv, N);
    gcn_deg_count<<<blocks(E), B, 0, stream>>>(dstp, E, dinv);
    gcn_dinv<<<blocks(N), B, 0, stream>>>(dinv, N);

    const int  mtiles = (int)((N + 15) / 16);
    const long eth    = (long)E * 32;
    const long NH4    = NH / 4;

    struct { const float* act; float* agg; int relu; } L[3] = {
        { x,    bufA, 1 },
        { bufA, bufB, 1 },
        { bufB, bufA, 0 },
    };

    for (int l = 0; l < 3; ++l) {
        gcn_cvt_f16<<<blocks(NH4), B, 0, stream>>>(L[l].act, actH, NH4);
        gcn_wt_cvt<<<HID, HID, 0, stream>>>(Ws[l], Wt);
        gcn_gemm_wmma<<<mtiles, B, 0, stream>>>(actH, Wt, hbuf, (int)N);
        gcn_zero_f32x4<<<blocks(NH4), B, 0, stream>>>((float4*)L[l].agg, NH4);
        gcn_scatter<<<blocks(eth), B, 0, stream>>>(hbuf, src, dstp, dinv, L[l].agg, E);
        gcn_finalize<<<blocks(NH4), B, 0, stream>>>(L[l].agg, hbuf, dinv, bs[l], L[l].relu, NH4);
    }

    gcn_decode<<<blocks(eth), B, 0, stream>>>(bufA, src, dstp, (float*)d_out, E);
}
